// AttentionBlock_8778913153473
// MI455X (gfx1250) — matrix operand order repaired and checked by reading, not compiled, not run on hardware
//
#include <hip/hip_runtime.h>
#include <hip/hip_bf16.h>

typedef _Float16 v16h __attribute__((ext_vector_type(16)));
typedef _Float16 h2   __attribute__((ext_vector_type(2)));
typedef float    v8f  __attribute__((ext_vector_type(8)));

static constexpr int C_DIM = 512;
static constexpr int G_DIM = 32;
static constexpr int NHEAD = 8;
static constexpr int HSZ   = 64;
static constexpr int BATCH = 2;
static constexpr int NTOK  = 4096;   // 64*64

#define WMMA_F16(a, b, c) \
  __builtin_amdgcn_wmma_f32_16x16x32_f16(false, (a), false, (b), (short)0, (c), false, false)

// ---------------------------------------------------------------------------
// A-fragment loader: 16x32 f16 A tile in the contiguous k order this file also uses for B (wave32).
// lane L: M = L&15 ; half i: K = 16*(L/16) + 2*i + {0,1}
// K-pairs are contiguous -> 8 dword gathers per lane (compiler merges to b128).
// ---------------------------------------------------------------------------
__device__ inline v16h load_a_frag(const _Float16* __restrict__ base, int stride,
                                   int row, int k0, int lane) {
  v16h a;
  const _Float16* p = base + (size_t)row * stride + k0 + 16 * (lane >> 4);
#pragma unroll
  for (int i = 0; i < 8; ++i) {
    int k = 2 * i;
    h2 v = *(const h2*)(p + k);
    a[2 * i]     = v.x;
    a[2 * i + 1] = v.y;
  }
  return a;
}

// A-fragment reload from per-wave LDS scratch (16 rows x 32 cols f16)
__device__ inline v16h load_p_frag(const _Float16* pw, int hi, int lm) {
  v16h a;
  const _Float16* pp = pw + lm * 32 + 16 * hi;
#pragma unroll
  for (int i = 0; i < 8; ++i) {
    int k = 2 * i;
    h2 v = *(const h2*)(pp + k);
    a[2 * i]     = v.x;
    a[2 * i + 1] = v.y;
  }
  return a;
}

// ---------------------------------------------------------------------------
// Online-softmax update for one 16x32 score block (two 16x16 C-layout tiles).
// Rows live in 16-lane halves: row = r + 8*(lane/16), col = lane&15.
// Scales scores by 1/sqrt(64), updates running max/sum, rescales O, and
// spills probabilities (f16) to per-wave LDS for A-fragment reload.
// ---------------------------------------------------------------------------
__device__ inline void softmax_update(v8f& s0, v8f& s1, float mrow[8], float lrow[8],
                                      v8f o[4], _Float16* pw, int hi, int lm) {
#pragma unroll
  for (int r = 0; r < 8; ++r) {
    float a  = s0[r] * 0.125f;
    float bb = s1[r] * 0.125f;
    float v = fmaxf(a, bb);
#pragma unroll
    for (int off = 1; off < 16; off <<= 1) v = fmaxf(v, __shfl_xor(v, off, 32));
    float nm    = fmaxf(mrow[r], v);
    float alpha = __expf(mrow[r] - nm);
    mrow[r]     = nm;
    float p0 = __expf(a - nm);
    float p1 = __expf(bb - nm);
    float sv = p0 + p1;
#pragma unroll
    for (int off = 1; off < 16; off <<= 1) sv += __shfl_xor(sv, off, 32);
    lrow[r] = alpha * lrow[r] + sv;
    o[0][r] *= alpha; o[1][r] *= alpha; o[2][r] *= alpha; o[3][r] *= alpha;
    pw[(r + 8 * hi) * 32 + lm]      = (_Float16)p0;
    pw[(r + 8 * hi) * 32 + 16 + lm] = (_Float16)p1;
  }
}

// ---------------------------------------------------------------------------
// Weight fp32 -> f16 conversion
// ---------------------------------------------------------------------------
__global__ __launch_bounds__(256) void wconv_kernel(const float* __restrict__ src,
                                                    _Float16* __restrict__ dst, int n) {
  int i = blockIdx.x * blockDim.x + threadIdx.x;
  if (i < n) dst[i] = (_Float16)src[i];
}

// ---------------------------------------------------------------------------
// GroupNorm: one block per (b, g). 16 channels x 4096 pixels contiguous.
// Writes normalized activations as f16 in [B, N, C] (token-major) layout.
// ---------------------------------------------------------------------------
__global__ __launch_bounds__(256) void gn_kernel(const float* __restrict__ x,
                                                 const float* __restrict__ w,
                                                 const float* __restrict__ bgn,
                                                 _Float16* __restrict__ t) {
  const int bg  = blockIdx.x;            // b*G + g
  const int b   = bg / G_DIM;
  const int g   = bg % G_DIM;
  const int tid = threadIdx.x;
  const size_t base = (size_t)bg * 16 * NTOK;
  const int    nel  = 16 * NTOK;

  float s = 0.f, ss = 0.f;
  for (int i = tid; i < nel; i += 256) {
    float v = x[base + i];
    s += v;
    ss += v * v;
  }
  __shared__ float rs[256], rss[256];
  rs[tid] = s; rss[tid] = ss;
  __syncthreads();
  for (int st = 128; st > 0; st >>= 1) {
    if (tid < st) { rs[tid] += rs[tid + st]; rss[tid] += rss[tid + st]; }
    __syncthreads();
  }
  const float mean = rs[0] * (1.0f / 65536.0f);
  const float var  = rss[0] * (1.0f / 65536.0f) - mean * mean;
  const float rstd = rsqrtf(var + 1e-5f);

  for (int i = tid; i < nel; i += 256) {
    int cc = i >> 12;              // channel within group
    int n  = i & (NTOK - 1);       // pixel
    int c  = g * 16 + cc;
    float v = (x[base + i] - mean) * rstd * w[c] + bgn[c];
    t[((size_t)b * NTOK + n) * C_DIM + c] = (_Float16)v;
  }
}

// ---------------------------------------------------------------------------
// WMMA GEMM: out[row, col] = sum_k A[row,k] * W[col,k] + bias[col]
//   A : f16 [B, N, C] ; W : f16 [C, C] row = output channel
//   mode 0: out16 = Q/K as [B, NH, N, HS]
//   mode 1: out16 = V  as [B, NH, HS, N]   (transposed for P*V fragments)
//   mode 2: out32 = proj output + residual, fp32 [B, C, N]
// Block = 8 waves, each wave computes a 32x32 tile (2x2 WMMA accumulators).
// Block tile: 64 rows x 128 cols.  Grid = (B*N/64) * (C/128).
// ---------------------------------------------------------------------------
__global__ __launch_bounds__(256) void gemm_kernel(const _Float16* __restrict__ A,
                                                   const _Float16* __restrict__ W,
                                                   const float* __restrict__ bias,
                                                   _Float16* __restrict__ out16,
                                                   float* __restrict__ out32,
                                                   const float* __restrict__ resid,
                                                   int mode) {
  const int MT    = BATCH * (NTOK / 64);            // 128 row-blocks
  const int bm    = blockIdx.x % MT;
  const int bn    = blockIdx.x / MT;
  const int b     = bm / (NTOK / 64);
  const int mbase = (bm % (NTOK / 64)) * 64;
  const int nbase = bn * 128;
  const int lane  = threadIdx.x & 31;
  const int wid   = threadIdx.x >> 5;
  const int m0    = mbase + (wid & 1) * 32;
  const int n0    = nbase + (wid >> 1) * 32;
  const int hi    = lane >> 4;
  const int lm    = lane & 15;

  const _Float16* Ab = A + (size_t)b * NTOK * C_DIM;
  const int col0 = n0 + lm;
  const int col1 = col0 + 16;

  v8f c00 = {}, c01 = {}, c10 = {}, c11 = {};
  for (int kk = 0; kk < C_DIM; kk += 32) {
    __builtin_prefetch((const void*)(W + (size_t)col0 * C_DIM + kk + 64), 0, 0);
    v16h a0 = load_a_frag(Ab, C_DIM, m0 + lm,      kk, lane);
    v16h a1 = load_a_frag(Ab, C_DIM, m0 + 16 + lm, kk, lane);
    // B fragment: lane holds 16 contiguous K halves of row `col` of W
    v16h b0 = *(const v16h*)(W + (size_t)col0 * C_DIM + kk + 16 * hi);
    v16h b1 = *(const v16h*)(W + (size_t)col1 * C_DIM + kk + 16 * hi);
    c00 = WMMA_F16(a0, b0, c00);
    c01 = WMMA_F16(a0, b1, c01);
    c10 = WMMA_F16(a1, b0, c10);
    c11 = WMMA_F16(a1, b1, c11);
  }

  const float bv0 = bias[col0];
  const float bv1 = bias[col1];

#pragma unroll
  for (int r = 0; r < 8; ++r) {
    int row0 = m0 + r + 8 * hi;
    int row1 = row0 + 16;
    float v00 = c00[r] + bv0, v01 = c01[r] + bv1;
    float v10 = c10[r] + bv0, v11 = c11[r] + bv1;
    if (mode == 2) {
      size_t i00 = ((size_t)b * C_DIM + col0) * NTOK + row0;
      size_t i01 = ((size_t)b * C_DIM + col1) * NTOK + row0;
      size_t i10 = ((size_t)b * C_DIM + col0) * NTOK + row1;
      size_t i11 = ((size_t)b * C_DIM + col1) * NTOK + row1;
      out32[i00] = v00 + resid[i00];
      out32[i01] = v01 + resid[i01];
      out32[i10] = v10 + resid[i10];
      out32[i11] = v11 + resid[i11];
    } else {
      int h0 = col0 >> 6, d0 = col0 & 63;
      int h1 = col1 >> 6, d1 = col1 & 63;
      size_t i00, i01, i10, i11;
      if (mode == 0) {
        i00 = (((size_t)b * NHEAD + h0) * NTOK + row0) * HSZ + d0;
        i01 = (((size_t)b * NHEAD + h1) * NTOK + row0) * HSZ + d1;
        i10 = (((size_t)b * NHEAD + h0) * NTOK + row1) * HSZ + d0;
        i11 = (((size_t)b * NHEAD + h1) * NTOK + row1) * HSZ + d1;
      } else {
        i00 = (((size_t)b * NHEAD + h0) * HSZ + d0) * NTOK + row0;
        i01 = (((size_t)b * NHEAD + h1) * HSZ + d1) * NTOK + row0;
        i10 = (((size_t)b * NHEAD + h0) * HSZ + d0) * NTOK + row1;
        i11 = (((size_t)b * NHEAD + h1) * HSZ + d1) * NTOK + row1;
      }
      out16[i00] = (_Float16)v00;
      out16[i01] = (_Float16)v01;
      out16[i10] = (_Float16)v10;
      out16[i11] = (_Float16)v11;
    }
  }
}

// ---------------------------------------------------------------------------
// Flash attention: one wave = 32 query rows (two 16-row M tiles), so each K/V
// fragment load feeds two WMMAs (16 WMMA per 8 v16h loads per key block).
// Block = 4 waves = 128 query rows.  Grid = B * NH * (N/128) = 512 blocks.
// ---------------------------------------------------------------------------
__global__ __launch_bounds__(128) void attn_kernel(const _Float16* __restrict__ Q,
                                                   const _Float16* __restrict__ K,
                                                   const _Float16* __restrict__ Vt,
                                                   _Float16* __restrict__ att) {
  const int qt   = blockIdx.x % (NTOK / 128);
  const int bh   = blockIdx.x / (NTOK / 128);   // b*NH + h
  const int b    = bh / NHEAD;
  const int h    = bh % NHEAD;
  const int lane = threadIdx.x & 31;
  const int wid  = threadIdx.x >> 5;
  const int hi   = lane >> 4;
  const int lm   = lane & 15;
  const int qbase = qt * 128 + wid * 32;        // this wave's first query row

  const _Float16* Qb = Q  + (size_t)bh * NTOK * HSZ;
  const _Float16* Kb = K  + (size_t)bh * NTOK * HSZ;
  const _Float16* Vb = Vt + (size_t)bh * HSZ * NTOK;

  __shared__ __align__(64) _Float16 plds[4][2][16 * 32];
  _Float16* pwA = &plds[wid][0][0];
  _Float16* pwB = &plds[wid][1][0];

  // Q fragments (held for the whole loop): 2 M tiles x 64 d
  v16h aq0A = load_a_frag(Qb, HSZ, qbase + lm,      0,  lane);
  v16h aq1A = load_a_frag(Qb, HSZ, qbase + lm,      32, lane);
  v16h aq0B = load_a_frag(Qb, HSZ, qbase + 16 + lm, 0,  lane);
  v16h aq1B = load_a_frag(Qb, HSZ, qbase + 16 + lm, 32, lane);

  float mA[8], lA[8], mB[8], lB[8];
#pragma unroll
  for (int r = 0; r < 8; ++r) { mA[r] = mB[r] = -3.0e38f; lA[r] = lB[r] = 0.0f; }
  v8f oA[4] = {}, oB[4] = {};

  for (int kb = 0; kb < NTOK; kb += 32) {
    // ---- K fragments (shared by both M tiles): 32 keys x 64 d, 2 k-steps
    v16h bk00 = *(const v16h*)(Kb + (size_t)(kb + lm)      * HSZ      + 16 * hi);
    v16h bk01 = *(const v16h*)(Kb + (size_t)(kb + lm)      * HSZ + 32 + 16 * hi);
    v16h bk10 = *(const v16h*)(Kb + (size_t)(kb + 16 + lm) * HSZ      + 16 * hi);
    v16h bk11 = *(const v16h*)(Kb + (size_t)(kb + 16 + lm) * HSZ + 32 + 16 * hi);

    v8f sA0 = {}, sA1 = {}, sB0 = {}, sB1 = {};
    sA0 = WMMA_F16(aq0A, bk00, sA0);
    sA0 = WMMA_F16(aq1A, bk01, sA0);
    sA1 = WMMA_F16(aq0A, bk10, sA1);
    sA1 = WMMA_F16(aq1A, bk11, sA1);
    sB0 = WMMA_F16(aq0B, bk00, sB0);
    sB0 = WMMA_F16(aq1B, bk01, sB0);
    sB1 = WMMA_F16(aq0B, bk10, sB1);
    sB1 = WMMA_F16(aq1B, bk11, sB1);

    // ---- online softmax for both M tiles
    softmax_update(sA0, sA1, mA, lA, oA, pwA, hi, lm);
    softmax_update(sB0, sB1, mB, lB, oB, pwB, hi, lm);

    // ---- reload P as A fragments (LDS is in-order within a wave)
    v16h apA = load_p_frag(pwA, hi, lm);
    v16h apB = load_p_frag(pwB, hi, lm);

    // ---- V fragments (shared by both M tiles); V stored [d, N] contiguous
    v16h bv0 = *(const v16h*)(Vb + (size_t)(0  + lm) * NTOK + kb + 16 * hi);
    v16h bv1 = *(const v16h*)(Vb + (size_t)(16 + lm) * NTOK + kb + 16 * hi);
    v16h bv2 = *(const v16h*)(Vb + (size_t)(32 + lm) * NTOK + kb + 16 * hi);
    v16h bv3 = *(const v16h*)(Vb + (size_t)(48 + lm) * NTOK + kb + 16 * hi);

    oA[0] = WMMA_F16(apA, bv0, oA[0]);
    oA[1] = WMMA_F16(apA, bv1, oA[1]);
    oA[2] = WMMA_F16(apA, bv2, oA[2]);
    oA[3] = WMMA_F16(apA, bv3, oA[3]);
    oB[0] = WMMA_F16(apB, bv0, oB[0]);
    oB[1] = WMMA_F16(apB, bv1, oB[1]);
    oB[2] = WMMA_F16(apB, bv2, oB[2]);
    oB[3] = WMMA_F16(apB, bv3, oB[3]);
  }

  // ---- normalize and store as f16 [B, N, C] for the projection GEMM
#pragma unroll
  for (int r = 0; r < 8; ++r) {
    float invA = 1.0f / lA[r];
    float invB = 1.0f / lB[r];
    int rowA = qbase + r + 8 * hi;
    int rowB = rowA + 16;
    size_t baseA = ((size_t)b * NTOK + rowA) * C_DIM + h * HSZ;
    size_t baseB = ((size_t)b * NTOK + rowB) * C_DIM + h * HSZ;
    att[baseA + 0  + lm] = (_Float16)(oA[0][r] * invA);
    att[baseA + 16 + lm] = (_Float16)(oA[1][r] * invA);
    att[baseA + 32 + lm] = (_Float16)(oA[2][r] * invA);
    att[baseA + 48 + lm] = (_Float16)(oA[3][r] * invA);
    att[baseB + 0  + lm] = (_Float16)(oB[0][r] * invB);
    att[baseB + 16 + lm] = (_Float16)(oB[1][r] * invB);
    att[baseB + 32 + lm] = (_Float16)(oB[2][r] * invB);
    att[baseB + 48 + lm] = (_Float16)(oB[3][r] * invB);
  }
}

// ---------------------------------------------------------------------------
extern "C" void kernel_launch(void* const* d_in, const int* in_sizes, int n_in,
                              void* d_out, int out_size, void* d_ws, size_t ws_size,
                              hipStream_t stream) {
  const float* x      = (const float*)d_in[0];
  const float* gn_w   = (const float*)d_in[1];
  const float* gn_b   = (const float*)d_in[2];
  const float* wq_w   = (const float*)d_in[3];
  const float* wq_b   = (const float*)d_in[4];
  const float* wk_w   = (const float*)d_in[5];
  const float* wk_b   = (const float*)d_in[6];
  const float* wv_w   = (const float*)d_in[7];
  const float* wv_b   = (const float*)d_in[8];
  const float* proj_w = (const float*)d_in[9];
  const float* proj_b = (const float*)d_in[10];

  char* ws = (char*)d_ws;
  size_t off = 0;
  auto alloc = [&](size_t bytes) -> void* {
    void* p = (void*)(ws + off);
    off += (bytes + 255) & ~(size_t)255;
    return p;
  };

  const size_t WELEM = (size_t)C_DIM * C_DIM;           // 262144
  const size_t TELEM = (size_t)BATCH * NTOK * C_DIM;    // 4194304

  _Float16* Wq  = (_Float16*)alloc(WELEM * 2);
  _Float16* Wk  = (_Float16*)alloc(WELEM * 2);
  _Float16* Wv  = (_Float16*)alloc(WELEM * 2);
  _Float16* Wp  = (_Float16*)alloc(WELEM * 2);
  _Float16* T   = (_Float16*)alloc(TELEM * 2);
  _Float16* Qh  = (_Float16*)alloc(TELEM * 2);
  _Float16* Kh  = (_Float16*)alloc(TELEM * 2);
  _Float16* Vth = (_Float16*)alloc(TELEM * 2);
  _Float16* Att = (_Float16*)alloc(TELEM * 2);

  const int wblk = (int)((WELEM + 255) / 256);
  wconv_kernel<<<wblk, 256, 0, stream>>>(wq_w,   Wq, (int)WELEM);
  wconv_kernel<<<wblk, 256, 0, stream>>>(wk_w,   Wk, (int)WELEM);
  wconv_kernel<<<wblk, 256, 0, stream>>>(wv_w,   Wv, (int)WELEM);
  wconv_kernel<<<wblk, 256, 0, stream>>>(proj_w, Wp, (int)WELEM);

  gn_kernel<<<BATCH * G_DIM, 256, 0, stream>>>(x, gn_w, gn_b, T);

  const int gemm_blocks = (BATCH * (NTOK / 64)) * (C_DIM / 128);  // 512
  gemm_kernel<<<gemm_blocks, 256, 0, stream>>>(T, Wq, wq_b, Qh,  nullptr, nullptr, 0);
  gemm_kernel<<<gemm_blocks, 256, 0, stream>>>(T, Wk, wk_b, Kh,  nullptr, nullptr, 0);
  gemm_kernel<<<gemm_blocks, 256, 0, stream>>>(T, Wv, wv_b, Vth, nullptr, nullptr, 1);

  attn_kernel<<<BATCH * NHEAD * (NTOK / 128), 128, 0, stream>>>(Qh, Kh, Vth, Att);

  gemm_kernel<<<gemm_blocks, 256, 0, stream>>>(Att, Wp, proj_b, nullptr,
                                               (float*)d_out, x, 2);
}
